// HierarchyCassification_58961311039683
// MI455X (gfx1250) — matrix operand order repaired and checked by reading, not compiled, not run on hardware
//
#include <hip/hip_runtime.h>
#include <hip/hip_bf16.h>

typedef __attribute__((ext_vector_type(16))) _Float16 v16h;
typedef __attribute__((ext_vector_type(8)))  _Float16 v8h;
typedef __attribute__((ext_vector_type(4)))  _Float16 v4h;
typedef __attribute__((ext_vector_type(8)))  float    v8f;

#define HH   768
#define FF   2048
#define VV   128
#define BB   8
#define SS   512
#define LL   12
#define NT   (BB * SS)        // 4096 tokens
#define LDQKV (3 * HH)        // 2304
#define VPAD 66               // padded LDS row stride (floats) for V chunks

// ---------------------------------------------------------------------------
// WMMA fragment loaders (contiguous k order, same for A and B)
// A fragment 16x32 (MxK): lane m = lane&15; elems 0..15 -> K=k0+16*hi+e.
// ---------------------------------------------------------------------------
__device__ __forceinline__ v16h load_a_frag_f32(const float* __restrict__ row,
                                                int k0, int hi) {
  v16h a;
  int kA = k0 + hi * 16;
#pragma unroll
  for (int e = 0; e < 8; ++e) a[e] = (_Float16)row[kA + e];
#pragma unroll
  for (int e = 0; e < 8; ++e) a[8 + e] = (_Float16)row[kA + 8 + e];
  return a;
}

// B fragment 32x16 (KxN) from W[N,K] row-major fp32 (B = W^T):
// lane n = lane&15; elems 0..15 -> K = k0 + 16*hi + e.
__device__ __forceinline__ v16h load_bT_frag_f32(const float* __restrict__ row,
                                                 int k0, int hi) {
  v16h b;
  int kb = k0 + hi * 16;
#pragma unroll
  for (int e = 0; e < 16; ++e) b[e] = (_Float16)row[kb + e];
  return b;
}

// Same, but from pre-converted f16 weights: two 16B vector loads per lane.
__device__ __forceinline__ v16h load_bT_frag_f16(const _Float16* __restrict__ row,
                                                 int k0, int hi) {
  const v8h* p = (const v8h*)(row + k0 + hi * 16);
  v8h lo = p[0], hv = p[1];
  v16h b;
#pragma unroll
  for (int e = 0; e < 8; ++e) { b[e] = lo[e]; b[8 + e] = hv[e]; }
  return b;
}

#define WMMA_F16(a, b, c) \
  __builtin_amdgcn_wmma_f32_16x16x32_f16(false, (a), false, (b), (short)0, (c), false, false)

// ---------------------------------------------------------------------------
// CDNA5 async-to-LDS helpers (ASYNCcnt-tracked copies, GVS addressing)
// ---------------------------------------------------------------------------
__device__ __forceinline__ void async_wait_le16() {
  asm volatile("s_wait_asynccnt 0x10" ::: "memory");
}
__device__ __forceinline__ void async_wait_0() {
  asm volatile("s_wait_asynccnt 0x0" ::: "memory");
}
__device__ __forceinline__ void ds_wait_0() {
  asm volatile("s_wait_dscnt 0x0" ::: "memory");
}

// Stage one 32-key x 64-dim fp32 V chunk into LDS (row stride VPAD floats).
// Each lane owns one key row: 16 x b128 = 256B, contiguous within each lane.
__device__ __forceinline__ void stage_v_chunk(const float* __restrict__ gbase,
                                              unsigned lds_base, int lane) {
  unsigned voff = (unsigned)lane * (LDQKV * 4);   // global: key stride
  unsigned loff = lds_base + (unsigned)lane * (VPAD * 4);
  unsigned long long sbase = (unsigned long long)(uintptr_t)gbase;
#pragma unroll
  for (int i = 0; i < 16; ++i) {
    asm volatile("global_load_async_to_lds_b128 %0, %1, %2"
                 :: "v"(loff), "v"(voff), "s"(sbase) : "memory");
    voff += 16;
    loff += 16;
  }
}

// ---------------------------------------------------------------------------
// Per-layer fp32 -> f16 weight conversion (bandwidth-trivial, once per use)
// ---------------------------------------------------------------------------
__global__ __launch_bounds__(256) void wcvt_kernel(const float* __restrict__ src,
                                                   _Float16* __restrict__ dst) {
  int i = (blockIdx.x * 256 + threadIdx.x) * 4;
  float4 v = *(const float4*)(src + i);
  v4h o;
  o[0] = (_Float16)v.x; o[1] = (_Float16)v.y;
  o[2] = (_Float16)v.z; o[3] = (_Float16)v.w;
  *(v4h*)(dst + i) = o;
}

// ---------------------------------------------------------------------------
// GEMM: C[M,N] = act(A[M,K] @ W[N,K]^T + bias), f16 weights, fp32 activations.
// 8 waves/block; each wave computes a 32x64 tile (2 A-frags x 4 B-frags,
// 8 WMMA per k-step; B reused 2x, A reused 4x). Block tile 128x128.
// ---------------------------------------------------------------------------
__global__ __launch_bounds__(256) void gemm_bias_kernel(
    const float* __restrict__ A, const _Float16* __restrict__ Wh,
    const float* __restrict__ bias, float* __restrict__ C,
    int M, int N, int K, int relu) {
  const int lane = threadIdx.x & 31;
  const int wave = threadIdx.x >> 5;
  const int mw = wave & 3, nw = wave >> 2;
  const int m0 = blockIdx.y * 128 + mw * 32;
  const int n0 = blockIdx.x * 128 + nw * 64;
  const int hi = lane >> 4;
  const int ln = lane & 15;
  const float*    arow0 = A  + (size_t)(m0 + ln) * K;
  const float*    arow1 = arow0 + (size_t)16 * K;
  const _Float16* wrow  = Wh + (size_t)(n0 + ln) * K;

  v8f acc[2][4] = {};
  for (int k0 = 0; k0 < K; k0 += 32) {
    v16h a0 = load_a_frag_f32(arow0, k0, hi);
    v16h a1 = load_a_frag_f32(arow1, k0, hi);
#pragma unroll
    for (int t = 0; t < 4; ++t) {
      v16h b = load_bT_frag_f16(wrow + (size_t)t * 16 * K, k0, hi);
      acc[0][t] = WMMA_F16(a0, b, acc[0][t]);
      acc[1][t] = WMMA_F16(a1, b, acc[1][t]);
    }
  }
#pragma unroll
  for (int u = 0; u < 2; ++u) {
#pragma unroll
    for (int t = 0; t < 4; ++t) {
      int n = n0 + t * 16 + ln;
      float bval = bias[n];
#pragma unroll
      for (int r = 0; r < 8; ++r) {
        int m = m0 + u * 16 + r + hi * 8;
        float v = acc[u][t][r] + bval;
        if (relu) v = fmaxf(v, 0.0f);
        C[(size_t)m * N + n] = v;
      }
    }
  }
}

// ---------------------------------------------------------------------------
// Attention: one wave per (query-tile-of-16, head, batch).
// scores (16x512) in LDS; shfl softmax; P@V with async double-buffered
// V staging into LDS (f32 chunks of 32 keys x 64 dims).
// ---------------------------------------------------------------------------
__global__ __launch_bounds__(32) void attn_kernel(const float* __restrict__ qkv,
                                                  float* __restrict__ o) {
  __shared__ float probs[16][SS + 4];
  __shared__ float vbuf[2][32 * VPAD];
  const int lane = threadIdx.x;
  const int qt = blockIdx.x, h = blockIdx.y, b = blockIdx.z;
  const int hi = lane >> 4;
  const int m = lane & 15;
  const size_t tok0 = (size_t)b * SS;

  // Q fragments (two K-chunks cover head dim 64)
  const float* qrow = qkv + (tok0 + qt * 16 + m) * LDQKV + h * 64;
  v16h aQ[2];
#pragma unroll
  for (int c = 0; c < 2; ++c) aQ[c] = load_a_frag_f32(qrow, c * 32, hi);

  // scores = (Q K^T) / 8  -> LDS
  for (int j = 0; j < SS / 16; ++j) {
    const float* krow = qkv + (tok0 + j * 16 + m) * LDQKV + HH + h * 64;
    v8f sc = {};
#pragma unroll
    for (int c = 0; c < 2; ++c) {
      v16h bK = load_bT_frag_f32(krow, c * 32, hi);
      sc = WMMA_F16(aQ[c], bK, sc);
    }
#pragma unroll
    for (int r = 0; r < 8; ++r)
      probs[r + hi * 8][j * 16 + m] = sc[r] * 0.125f;
  }
  __syncthreads();

  // row-wise softmax (wave-cooperative, 16 values/lane per row)
  for (int r = 0; r < 16; ++r) {
    float mx = -__builtin_inff();
#pragma unroll
    for (int i = 0; i < 16; ++i) mx = fmaxf(mx, probs[r][lane + i * 32]);
    for (int off = 16; off; off >>= 1) mx = fmaxf(mx, __shfl_xor(mx, off, 32));
    float ev[16], sum = 0.0f;
#pragma unroll
    for (int i = 0; i < 16; ++i) {
      ev[i] = __expf(probs[r][lane + i * 32] - mx);
      sum += ev[i];
    }
    for (int off = 16; off; off >>= 1) sum += __shfl_xor(sum, off, 32);
    float inv = 1.0f / sum;
#pragma unroll
    for (int i = 0; i < 16; ++i) probs[r][lane + i * 32] = ev[i] * inv;
  }
  __syncthreads();

  // O(16x64) = P(16x512) @ V(512x64), V staged via async-to-LDS (double buf)
  const float* vsrc = qkv + tok0 * LDQKV + 2 * HH + h * 64;
  unsigned vb_lds0 = (unsigned)(uintptr_t)&vbuf[0][0];
  unsigned vb_lds1 = (unsigned)(uintptr_t)&vbuf[1][0];
  stage_v_chunk(vsrc, vb_lds0, lane);

  v8f acc[4] = {};
  for (int kk = 0; kk < SS / 32; ++kk) {
    const int cur = kk & 1;
    if (kk + 1 < SS / 32) {
      ds_wait_0();  // LDS reads of the buffer we are about to overwrite
      stage_v_chunk(vsrc + (size_t)(kk + 1) * 32 * LDQKV,
                    cur ? vb_lds0 : vb_lds1, lane);
      async_wait_le16();   // current chunk complete; next may be in flight
    } else {
      async_wait_0();
    }
    const float* vb = cur ? &vbuf[1][0] : &vbuf[0][0];
    const int k0 = kk * 32;
    v16h aP;
    {
      int kA = k0 + hi * 16;
#pragma unroll
      for (int e = 0; e < 8; ++e) aP[e] = (_Float16)probs[m][kA + e];
#pragma unroll
      for (int e = 0; e < 8; ++e) aP[8 + e] = (_Float16)probs[m][kA + 8 + e];
    }
#pragma unroll
    for (int t = 0; t < 4; ++t) {
      v16h bV;
#pragma unroll
      for (int e = 0; e < 16; ++e)
        bV[e] = (_Float16)vb[(e + hi * 16) * VPAD + t * 16 + m];
      acc[t] = WMMA_F16(aP, bV, acc[t]);
    }
  }
#pragma unroll
  for (int t = 0; t < 4; ++t)
#pragma unroll
    for (int r = 0; r < 8; ++r)
      o[(tok0 + qt * 16 + r + hi * 8) * HH + h * 64 + t * 16 + m] = acc[t][r];
}

// ---------------------------------------------------------------------------
// out = LayerNorm(x + y) * g + b   (one wave per token; may write in place)
// ---------------------------------------------------------------------------
__global__ __launch_bounds__(32) void add_ln_kernel(
    const float* __restrict__ x, const float* __restrict__ y,
    const float* __restrict__ g, const float* __restrict__ beta,
    float* __restrict__ out) {
  const int t = blockIdx.x, lane = threadIdx.x;
  const float* xr = x + (size_t)t * HH;
  const float* yr = y + (size_t)t * HH;
  float v[HH / 32];
  float s = 0.0f;
#pragma unroll
  for (int i = 0; i < HH / 32; ++i) {
    v[i] = xr[lane + i * 32] + yr[lane + i * 32];
    s += v[i];
  }
  for (int off = 16; off; off >>= 1) s += __shfl_xor(s, off, 32);
  float mu = s * (1.0f / HH);
  float q = 0.0f;
#pragma unroll
  for (int i = 0; i < HH / 32; ++i) { float d = v[i] - mu; q += d * d; }
  for (int off = 16; off; off >>= 1) q += __shfl_xor(q, off, 32);
  float inv = rsqrtf(q * (1.0f / HH) + 1e-5f);
  float* orow = out + (size_t)t * HH;
#pragma unroll
  for (int i = 0; i < HH / 32; ++i) {
    int d = lane + i * 32;
    orow[d] = g[d] * (v[i] - mu) * inv + beta[d];
  }
}

// ---------------------------------------------------------------------------
// Embedding sum
// ---------------------------------------------------------------------------
__global__ __launch_bounds__(256) void embed_kernel(
    const int* __restrict__ lid, const int* __restrict__ sid,
    const float* __restrict__ be, const float* __restrict__ se,
    float* __restrict__ x) {
  const int t = blockIdx.x;
  const int l = lid[t], s = sid[t];
  for (int d = threadIdx.x; d < HH; d += 256)
    x[(size_t)t * HH + d] = be[(size_t)l * HH + d] + se[(size_t)s * HH + d];
}

// ---------------------------------------------------------------------------
// Argmax over 128 logits per token (first-occurrence ties, like jnp.argmax)
// ---------------------------------------------------------------------------
__global__ __launch_bounds__(32) void argmax_kernel(const float* __restrict__ logits,
                                                    int* __restrict__ ix) {
  const int t = blockIdx.x, lane = threadIdx.x;
  const float* row = logits + (size_t)t * VV;
  float bv = -__builtin_inff();
  int bi = 0;
#pragma unroll
  for (int i = 0; i < VV / 32; ++i) {
    int idx = lane + i * 32;
    float v = row[idx];
    if (v > bv || (v == bv && idx < bi)) { bv = v; bi = idx; }
  }
  for (int off = 16; off; off >>= 1) {
    float ov = __shfl_xor(bv, off, 32);
    int   oi = __shfl_xor(bi, off, 32);
    if (ov > bv || (ov == bv && oi < bi)) { bv = ov; bi = oi; }
  }
  if (lane == 0) ix[t] = bi;
}

// ---------------------------------------------------------------------------
// Routed per-token vec-mat: out[t,v] = x[t,:] . W[ix[t],:,v] + bias[ix[t],v]
// W (50 MB) is L2-resident; consecutive threads read consecutive v (coalesced).
// ---------------------------------------------------------------------------
__global__ __launch_bounds__(128) void routed_kernel(
    const float* __restrict__ x, const float* __restrict__ W,
    const float* __restrict__ bias, const int* __restrict__ ix,
    float* __restrict__ out) {
  __shared__ float xs[HH];
  const int t = blockIdx.x, v = threadIdx.x;
  const float* xr = x + (size_t)t * HH;
  for (int i = v; i < HH; i += 128) xs[i] = xr[i];
  __syncthreads();
  const int lv = ix[t];
  const float* Wp = W + (size_t)lv * HH * VV + v;
  float acc = bias[(size_t)lv * VV + v];
#pragma unroll 8
  for (int hh = 0; hh < HH; ++hh) acc = fmaf(xs[hh], Wp[(size_t)hh * VV], acc);
  out[(size_t)t * VV + v] = acc;
}

// ---------------------------------------------------------------------------
// Host driver
// ---------------------------------------------------------------------------
static inline void run_gemm(const float* A, const float* W32, _Float16* wbuf,
                            const float* bias, float* C, int M, int N, int K,
                            int relu, hipStream_t stream) {
  wcvt_kernel<<<(N * K) / 1024, 256, 0, stream>>>(W32, wbuf);
  gemm_bias_kernel<<<dim3(N / 128, M / 128), 256, 0, stream>>>(
      A, wbuf, bias, C, M, N, K, relu);
}

extern "C" void kernel_launch(void* const* d_in, const int* in_sizes, int n_in,
                              void* d_out, int out_size, void* d_ws, size_t ws_size,
                              hipStream_t stream) {
  const int*   level_ids = (const int*)d_in[0];
  const int*   sub_ids   = (const int*)d_in[1];
  const float* bin_emb   = (const float*)d_in[2];
  const float* sub_emb   = (const float*)d_in[3];
  const float* Wqkv      = (const float*)d_in[4];
  const float* bqkv      = (const float*)d_in[5];
  const float* Wo        = (const float*)d_in[6];
  const float* bo        = (const float*)d_in[7];
  const float* ln1_g     = (const float*)d_in[8];
  const float* ln1_b     = (const float*)d_in[9];
  const float* W1        = (const float*)d_in[10];
  const float* b1        = (const float*)d_in[11];
  const float* W2        = (const float*)d_in[12];
  const float* b2        = (const float*)d_in[13];
  const float* ln2_g     = (const float*)d_in[14];
  const float* ln2_b     = (const float*)d_in[15];
  const float* W_lm      = (const float*)d_in[16];
  const float* b_lm      = (const float*)d_in[17];
  const float* Wrt       = (const float*)d_in[18];
  const float* brt       = (const float*)d_in[19];

  // Workspace layout (~79 MB). FFN hidden reuses the QKV buffer;
  // per-layer f16 weights go in a small reusable buffer (L2-resident).
  float* buf_x   = (float*)d_ws;                       // [4096, 768]
  float* buf_qkv = buf_x   + (size_t)NT * HH;          // [4096, 2304] / [4096, 2048]
  float* buf_a   = buf_qkv + (size_t)NT * LDQKV;       // [4096, 768]
  float* buf_b   = buf_a   + (size_t)NT * HH;          // [4096, 768]
  int*   ix      = (int*)(buf_b + (size_t)NT * HH);    // [4096]
  _Float16* wbuf = (_Float16*)(ix + NT);               // [<= 2304*768] f16

  float* out0 = (float*)d_out;            // logits  [4096, 128]
  float* out1 = out0 + (size_t)NT * VV;   // routed  [4096, 128]

  embed_kernel<<<NT, 256, 0, stream>>>(level_ids, sub_ids, bin_emb, sub_emb, buf_x);

  for (int l = 0; l < LL; ++l) {
    const float* Wqkv_l = Wqkv + (size_t)l * 3 * HH * HH;
    const float* bqkv_l = bqkv + (size_t)l * 3 * HH;
    const float* Wo_l   = Wo   + (size_t)l * HH * HH;
    const float* bo_l   = bo   + (size_t)l * HH;
    const float* g1_l   = ln1_g + (size_t)l * HH;
    const float* be1_l  = ln1_b + (size_t)l * HH;
    const float* W1_l   = W1   + (size_t)l * FF * HH;
    const float* b1_l   = b1   + (size_t)l * FF;
    const float* W2_l   = W2   + (size_t)l * HH * FF;
    const float* b2_l   = b2   + (size_t)l * HH;
    const float* g2_l   = ln2_g + (size_t)l * HH;
    const float* be2_l  = ln2_b + (size_t)l * HH;

    // qkv = x @ Wqkv^T + bqkv
    run_gemm(buf_x, Wqkv_l, wbuf, bqkv_l, buf_qkv, NT, 3 * HH, HH, 0, stream);
    // attention -> buf_a
    attn_kernel<<<dim3(SS / 16, 12, BB), 32, 0, stream>>>(buf_qkv, buf_a);
    // o = attn_out @ Wo^T + bo
    run_gemm(buf_a, Wo_l, wbuf, bo_l, buf_b, NT, HH, HH, 0, stream);
    // x = LN(x + o)
    add_ln_kernel<<<NT, 32, 0, stream>>>(buf_x, buf_b, g1_l, be1_l, buf_x);
    // f = relu(x @ W1^T + b1)  (reuse qkv buffer)
    run_gemm(buf_x, W1_l, wbuf, b1_l, buf_qkv, NT, FF, HH, 1, stream);
    // f2 = f @ W2^T + b2
    run_gemm(buf_qkv, W2_l, wbuf, b2_l, buf_b, NT, HH, FF, 0, stream);
    // x = LN(x + f2)
    add_ln_kernel<<<NT, 32, 0, stream>>>(buf_x, buf_b, g2_l, be2_l, buf_x);
  }

  // logits = x @ W_lm^T + b_lm  -> first output
  run_gemm(buf_x, W_lm, wbuf, b_lm, out0, NT, VV, HH, 0, stream);
  // routing index
  argmax_kernel<<<NT, 32, 0, stream>>>(out0, ix);
  // routed vec-mat -> second output
  routed_kernel<<<NT, 128, 0, stream>>>(buf_x, Wrt, brt, ix, out1);
}
